// BLBlock_31190052503719
// MI455X (gfx1250) — matrix operand order repaired and checked by reading, not compiled, not run on hardware
//
#include <hip/hip_runtime.h>
#include <hip/hip_bf16.h>

// ---------------------------------------------------------------------------
// Mamba-style vision block for MI455X (gfx1250, wave32, WMMA).
// 5 GEMMs on v_wmma_f32_16x16x32_f16, register-blocked 32x64 per wave,
// fused epilogues; async global->LDS staging for the input-LN transpose.
// ---------------------------------------------------------------------------

typedef __attribute__((ext_vector_type(16))) _Float16 v16h;
typedef __attribute__((ext_vector_type(8)))  _Float16 v8h;
typedef __attribute__((ext_vector_type(8)))  float    v8f;

#define NB      8          // batch
#define LSEQ    4096       // H*W
#define MROWS   (NB*LSEQ)  // 32768
#define DMODEL  256
#define DINNER  512
#define DSTATE  16
#define DTRANK  16

#if defined(__HIP_DEVICE_COMPILE__) && \
    __has_builtin(__builtin_amdgcn_global_load_async_to_lds_b32) && \
    __has_builtin(__builtin_amdgcn_s_wait_asynccnt)
#define USE_ASYNC_LDS 1
#else
#define USE_ASYNC_LDS 0
#endif

// ----------------------------- f32 -> f16 conversion ------------------------
__global__ void cvt_f16_kernel(const float* __restrict__ in, _Float16* __restrict__ out, int n) {
    int i = blockIdx.x * blockDim.x + threadIdx.x;
    if (i < n) out[i] = (_Float16)in[i];
}

// ----------------------------- input LayerNorm (with NCHW->NLC transpose) ---
// x: (B, 256, 4096) f32 -> xs16: (B*4096, 256) f16, LN over C (eps 1e-6)
__global__ __launch_bounds__(256) void ln_in_kernel(const float* __restrict__ x,
                                                    const float* __restrict__ w,
                                                    const float* __restrict__ b,
                                                    _Float16* __restrict__ xs16) {
    __shared__ float tile[32][DMODEL + 1];
    __shared__ float red[2][32][8];
    const int bb = blockIdx.x >> 7;            // 128 blocks per batch
    const int t0 = (blockIdx.x & 127) << 5;    // 32 rows per block
    const int tx = threadIdx.x & 31;           // t within tile
    const int ty = threadIdx.x >> 5;           // c group
    const float* xb = x + ((size_t)bb << 20);  // b * 256 * 4096

#if USE_ASYNC_LDS
    // async global -> LDS staging (ASYNCcnt-tracked, no VGPR round trip)
    for (int c = ty; c < DMODEL; c += 8) {
        float* gp = const_cast<float*>(xb + ((size_t)c << 12) + t0 + tx);
        __builtin_amdgcn_global_load_async_to_lds_b32(
            (__attribute__((address_space(1))) int*)gp,
            (__attribute__((address_space(3))) int*)&tile[tx][c],
            0, 0);
    }
    __builtin_amdgcn_s_wait_asynccnt(0);
    __syncthreads();
#else
    for (int c = ty; c < DMODEL; c += 8)
        tile[tx][c] = xb[((size_t)c << 12) + t0 + tx];
    __syncthreads();
#endif

    float s = 0.f, s2 = 0.f;
    for (int c = ty; c < DMODEL; c += 8) { float v = tile[tx][c]; s += v; s2 += v * v; }
    red[0][tx][ty] = s; red[1][tx][ty] = s2;
    __syncthreads();
    float mu = 0.f, m2 = 0.f;
    #pragma unroll
    for (int j = 0; j < 8; ++j) { mu += red[0][tx][j]; m2 += red[1][tx][j]; }
    mu *= (1.f / DMODEL);
    m2 = m2 * (1.f / DMODEL) - mu * mu;
    const float rs = rsqrtf(m2 + 1e-6f);
    _Float16* orow = xs16 + (((size_t)bb << 12) + t0) * DMODEL;
    for (int c = ty; c < DMODEL; c += 8)
        orow[(size_t)tx * DMODEL + c] = (_Float16)((tile[tx][c] - mu) * rs * w[c] + b[c]);
}

// ----------------------------- WMMA GEMM: Y = A(MxK,f16) @ W(NxK,f16)^T -----
// Register-blocked: each wave computes a (16*MBLK) x (16*NBLK) output block,
// 8 v_wmma per K-step from 12 b128 loads (A fragments reused across NBLK,
// B fragments across MBLK).  KK compile-time -> fully unrolled wmma chain.
// MODE 0: in-proj.  n<512 -> out_f32 (xm_pre);  n>=512 -> silu -> out_f16 (z16)
// MODE 1: x-proj.   plain f32 store (dbl, N=48, NBLK=3)
// MODE 2: out-proj. c*beta[n] + x(b,n,t) residual -> yhw f32 (M x 256)
// MODE 3: fc1.      +bias, leaky-relu(0.01) -> f16
// MODE 4: fc2.      +bias + yhw residual, transposed store -> d_out (B,256,L)
template<int MODE, int KK, int NBLK>
__global__ __launch_bounds__(256) void wmma_gemm_kernel(
    const _Float16* __restrict__ A, const _Float16* __restrict__ W, int N,
    float* __restrict__ out_f32, _Float16* __restrict__ out_f16,
    const float* __restrict__ bias, const float* __restrict__ beta,
    const float* __restrict__ resid)
{
    constexpr int MBLK = 2;
    const int lane   = threadIdx.x & 31;
    const int wave   = threadIdx.x >> 5;
    const int gn_cnt = N / (16 * NBLK);
    const int gid    = blockIdx.x * 8 + wave;
    const int gm     = gid / gn_cnt;
    const int gn     = gid - gm * gn_cnt;
    const int m0     = gm * (16 * MBLK);
    const int n0     = gn * (16 * NBLK);
    const int half   = lane >> 4;    // 0 or 1
    const int r      = lane & 15;

    // A 16x32 f16 fragment: lane<16 holds K k0..k0+15
    //                       lane>=16 holds K k0+16..k0+31
    const _Float16* Ar[MBLK];
    #pragma unroll
    for (int mb = 0; mb < MBLK; ++mb)
        Ar[mb] = A + (size_t)(m0 + mb * 16 + r) * KK + half * 16;
    // B 32x16 f16 fragment: lane<16 col n+r K k0..k0+15; lane>=16 K k0+16..31
    const _Float16* Wr[NBLK];
    #pragma unroll
    for (int nb = 0; nb < NBLK; ++nb)
        Wr[nb] = W + (size_t)(n0 + nb * 16 + r) * KK + half * 16;

    v8f acc[MBLK][NBLK];
    #pragma unroll
    for (int mb = 0; mb < MBLK; ++mb)
        #pragma unroll
        for (int nb = 0; nb < NBLK; ++nb)
            acc[mb][nb] = (v8f){0.f,0.f,0.f,0.f,0.f,0.f,0.f,0.f};

    #pragma unroll
    for (int k0 = 0; k0 < KK; k0 += 32) {
        if (k0 + 128 < KK) {
            __builtin_prefetch(Ar[0] + k0 + 128, 0, 1);
            __builtin_prefetch(Ar[1] + k0 + 128, 0, 1);
        }
        v16h a[MBLK];
        #pragma unroll
        for (int mb = 0; mb < MBLK; ++mb) {
            v8h lo = *(const v8h*)(Ar[mb] + k0);
            v8h hi = *(const v8h*)(Ar[mb] + k0 + 8);
            a[mb]  = __builtin_shufflevector(lo, hi,
                        0,1,2,3,4,5,6,7,8,9,10,11,12,13,14,15);
        }
        #pragma unroll
        for (int nb = 0; nb < NBLK; ++nb) {
            v16h b = *(const v16h*)(Wr[nb] + k0);
            #pragma unroll
            for (int mb = 0; mb < MBLK; ++mb)
                acc[mb][nb] = __builtin_amdgcn_wmma_f32_16x16x32_f16(
                    false, a[mb], false, b, (short)0, acc[mb][nb], false, false);
        }
    }

    #pragma unroll
    for (int mb = 0; mb < MBLK; ++mb) {
        const int mrow0 = m0 + mb * 16 + half * 8;   // first of 8 output rows
        #pragma unroll
        for (int nb = 0; nb < NBLK; ++nb) {
            const int cidx = n0 + nb * 16 + r;       // output column
            v8f c = acc[mb][nb];

            if (MODE == 0) {
                if (cidx < DINNER) {
                    #pragma unroll
                    for (int rr = 0; rr < 8; ++rr)
                        out_f32[(size_t)(mrow0 + rr) * DINNER + cidx] = c[rr];
                } else {
                    const int cz = cidx - DINNER;
                    #pragma unroll
                    for (int rr = 0; rr < 8; ++rr) {
                        float v = c[rr];
                        out_f16[(size_t)(mrow0 + rr) * DINNER + cz] =
                            (_Float16)(v / (1.f + __expf(-v)));   // silu(z)
                    }
                }
            } else if (MODE == 1) {
                #pragma unroll
                for (int rr = 0; rr < 8; ++rr)
                    out_f32[(size_t)(mrow0 + rr) * 48 + cidx] = c[rr];
            } else if (MODE == 2) {
                const float bt = beta[cidx];
                const int bidx = m0 >> 12;
                const float* xr = resid + ((size_t)(bidx * DMODEL + cidx) << 12);
                const int tt = (m0 & 4095) + mb * 16 + half * 8;
                #pragma unroll
                for (int rr = 0; rr < 8; ++rr)
                    out_f32[(size_t)(mrow0 + rr) * DMODEL + cidx] =
                        c[rr] * bt + xr[tt + rr];
            } else if (MODE == 3) {
                const float bb = bias[cidx];
                #pragma unroll
                for (int rr = 0; rr < 8; ++rr) {
                    float v = c[rr] + bb;
                    v = (v >= 0.f) ? v : 0.01f * v;               // leaky relu
                    out_f16[(size_t)(mrow0 + rr) * DINNER + cidx] = (_Float16)v;
                }
            } else {  // MODE 4: transpose store to (B, 256, L) + residual
                const float bb = bias[cidx];
                const int bidx = m0 >> 12;
                float* op = out_f32 + ((size_t)(bidx * DMODEL + cidx) << 12);
                const int tt = (m0 & 4095) + mb * 16 + half * 8;
                #pragma unroll
                for (int rr = 0; rr < 8; ++rr)
                    op[tt + rr] = c[rr] + bb +
                        resid[(size_t)(mrow0 + rr) * DMODEL + cidx];
            }
        }
    }
}

// ----------------------------- depthwise causal conv(k=4) + SiLU ------------
// xm_pre: (M, 512) f32 -> xm f32 (scan input) + xm16 (x-proj input)
__global__ __launch_bounds__(256) void conv_silu_kernel(
    const float* __restrict__ xm_pre, const float* __restrict__ cw,
    const float* __restrict__ cb, float* __restrict__ xm_f32,
    _Float16* __restrict__ xm16)
{
    const size_t idx = (size_t)blockIdx.x * 256 + threadIdx.x;  // over M*512
    const int d = (int)(idx & 511);
    const size_t m = idx >> 9;
    const int t = (int)(m & 4095);
    float acc = cb[d];
    #pragma unroll
    for (int j = 0; j < 4; ++j) {
        const int tt = t - 3 + j;
        if (tt >= 0)
            acc += cw[d * 4 + j] * xm_pre[(m - (size_t)(3 - j)) * DINNER + d];
    }
    const float sv = acc / (1.f + __expf(-acc));   // silu
    xm_f32[idx] = sv;
    xm16[idx]   = (_Float16)sv;
}

// ----------------------------- delta = softplus(dt @ w_dt^T + b_dt) ---------
__global__ __launch_bounds__(256) void delta_kernel(
    const float* __restrict__ dbl, const float* __restrict__ w_dt,
    const float* __restrict__ b_dt, float* __restrict__ delta)
{
    const size_t idx = (size_t)blockIdx.x * 256 + threadIdx.x;  // over M*512
    const int d = (int)(idx & 511);
    const size_t m = idx >> 9;
    const float* dr = dbl + m * 48;         // dt = dbl[:, 0:16]
    const float* wr = w_dt + d * DTRANK;
    float acc = b_dt[d];
    #pragma unroll
    for (int rr = 0; rr < DTRANK; ++rr) acc += dr[rr] * wr[rr];
    delta[idx] = (acc > 20.f) ? acc : log1pf(__expf(acc));
}

// ----------------------------- selective scan -------------------------------
// one lane per (b, d) channel; 16 f32 states in registers; B/C staged in LDS.
// epilogue: gate with pre-computed silu(z) and emit f16 for out-proj GEMM.
__global__ __launch_bounds__(128) void scan_kernel(
    const float* __restrict__ xm, const float* __restrict__ delta,
    const float* __restrict__ dbl, const float* __restrict__ a_log,
    const float* __restrict__ d_skip, const _Float16* __restrict__ z16,
    _Float16* __restrict__ ymul16)
{
    const int bb = blockIdx.x >> 2;                      // batch
    const int d  = ((blockIdx.x & 3) << 7) + threadIdx.x;
    float Av[DSTATE];
    #pragma unroll
    for (int n = 0; n < DSTATE; ++n) Av[n] = -__expf(a_log[d * DSTATE + n]);
    float h[DSTATE];
    #pragma unroll
    for (int n = 0; n < DSTATE; ++n) h[n] = 0.f;
    const float dsk = d_skip[d];

    __shared__ float bc[2 * DSTATE];
    const size_t mbase = (size_t)bb << 12;
    for (int t = 0; t < LSEQ; ++t) {
        const size_t m = mbase + t;
        __syncthreads();
        if (threadIdx.x < 2 * DSTATE)
            bc[threadIdx.x] = dbl[m * 48 + DTRANK + threadIdx.x];
        __syncthreads();
        const size_t idx = (m << 9) + d;
        const float dt = delta[idx];
        const float u  = xm[idx];
        const float du = dt * u;
        float y = 0.f;
        #pragma unroll
        for (int n = 0; n < DSTATE; ++n) {
            const float dA = __expf(dt * Av[n]);
            h[n] = dA * h[n] + du * bc[n];
            y += h[n] * bc[DSTATE + n];
        }
        const float zg = (float)z16[idx];                // silu(z) precomputed
        ymul16[idx] = (_Float16)((y + u * dsk) * zg);
    }
}

// ----------------------------- FFN LayerNorm (rows contiguous, eps 1e-5) ----
__global__ __launch_bounds__(256) void ln_ffn_kernel(
    const float* __restrict__ yhw, const float* __restrict__ w,
    const float* __restrict__ b, _Float16* __restrict__ out16)
{
    const int row  = blockIdx.x * 8 + (threadIdx.x >> 5);
    const int lane = threadIdx.x & 31;
    const float* rp = yhw + (size_t)row * DMODEL + lane * 8;
    float4 v0 = *(const float4*)(rp);
    float4 v1 = *(const float4*)(rp + 4);
    float s  = v0.x + v0.y + v0.z + v0.w + v1.x + v1.y + v1.z + v1.w;
    float s2 = v0.x*v0.x + v0.y*v0.y + v0.z*v0.z + v0.w*v0.w +
               v1.x*v1.x + v1.y*v1.y + v1.z*v1.z + v1.w*v1.w;
    #pragma unroll
    for (int msk = 16; msk >= 1; msk >>= 1) {
        s  += __shfl_xor(s,  msk, 32);
        s2 += __shfl_xor(s2, msk, 32);
    }
    const float mu = s * (1.f / DMODEL);
    const float rs = rsqrtf(s2 * (1.f / DMODEL) - mu * mu + 1e-5f);
    _Float16* op = out16 + (size_t)row * DMODEL + lane * 8;
    const float* wp = w + lane * 8;
    const float* bp = b + lane * 8;
    const float vv[8] = {v0.x, v0.y, v0.z, v0.w, v1.x, v1.y, v1.z, v1.w};
    #pragma unroll
    for (int j = 0; j < 8; ++j)
        op[j] = (_Float16)((vv[j] - mu) * rs * wp[j] + bp[j]);
}

// ---------------------------------------------------------------------------
extern "C" void kernel_launch(void* const* d_in, const int* in_sizes, int n_in,
                              void* d_out, int out_size, void* d_ws, size_t ws_size,
                              hipStream_t stream) {
    const float* x       = (const float*)d_in[0];
    const float* ln_in_w = (const float*)d_in[1];
    const float* ln_in_b = (const float*)d_in[2];
    const float* w_in    = (const float*)d_in[3];   // (1024, 256)
    const float* conv_w  = (const float*)d_in[4];   // (512, 1, 4)
    const float* conv_b  = (const float*)d_in[5];
    const float* w_x     = (const float*)d_in[6];   // (48, 512)
    const float* w_dt    = (const float*)d_in[7];   // (512, 16)
    const float* b_dt    = (const float*)d_in[8];
    const float* a_log   = (const float*)d_in[9];   // (512, 16)
    const float* d_skip  = (const float*)d_in[10];
    const float* w_out   = (const float*)d_in[11];  // (256, 512)
    const float* beta    = (const float*)d_in[12];  // (256)
    const float* ln_f_w  = (const float*)d_in[13];
    const float* ln_f_b  = (const float*)d_in[14];
    const float* fc1_w   = (const float*)d_in[15];  // (512, 256)
    const float* fc1_b   = (const float*)d_in[16];
    const float* fc2_w   = (const float*)d_in[17];  // (256, 512)
    const float* fc2_b   = (const float*)d_in[18];
    float* out = (float*)d_out;

    // ---- workspace carve-out (256B aligned) ----
    char* ws = (char*)d_ws;
    size_t off = 0;
    auto alloc = [&](size_t bytes) -> char* {
        char* p = ws + off;
        off = (off + bytes + 255) & ~size_t(255);
        return p;
    };
    _Float16* xs16    = (_Float16*)alloc((size_t)MROWS * DMODEL * 2);
    _Float16* w_in16  = (_Float16*)alloc((size_t)1024 * 256 * 2);
    _Float16* wx16    = (_Float16*)alloc((size_t)48 * 512 * 2);
    _Float16* wout16  = (_Float16*)alloc((size_t)256 * 512 * 2);
    _Float16* fc116   = (_Float16*)alloc((size_t)512 * 256 * 2);
    _Float16* fc216   = (_Float16*)alloc((size_t)256 * 512 * 2);
    float*    xm_pre  = (float*)   alloc((size_t)MROWS * DINNER * 4);  // reused as delta
    _Float16* z16     = (_Float16*)alloc((size_t)MROWS * DINNER * 2);
    float*    xm_f32  = (float*)   alloc((size_t)MROWS * DINNER * 4);
    _Float16* xm16    = (_Float16*)alloc((size_t)MROWS * DINNER * 2);
    float*    dbl     = (float*)   alloc((size_t)MROWS * 48 * 4);
    _Float16* ymul16  = (_Float16*)alloc((size_t)MROWS * DINNER * 2);
    float*    yhw     = (float*)   alloc((size_t)MROWS * DMODEL * 4);
    _Float16* ln16    = (_Float16*)alloc((size_t)MROWS * DMODEL * 2);
    _Float16* t116    = (_Float16*)alloc((size_t)MROWS * DINNER * 2);
    float*    deltaB  = xm_pre;   // xm_pre fully consumed by conv before delta

    // ---- 1. weights to f16 ----
    auto cvt = [&](const float* src, _Float16* dst, int n) {
        cvt_f16_kernel<<<(n + 255) / 256, 256, 0, stream>>>(src, dst, n);
    };
    cvt(w_in,  w_in16, 1024 * 256);
    cvt(w_x,   wx16,   48 * 512);
    cvt(w_out, wout16, 256 * 512);
    cvt(fc1_w, fc116,  512 * 256);
    cvt(fc2_w, fc216,  256 * 512);

    // ---- 2. input LN (+ transpose NCHW -> (M,256) f16) ----
    ln_in_kernel<<<NB * 128, 256, 0, stream>>>(x, ln_in_w, ln_in_b, xs16);

    // ---- 3. in-proj GEMM: (M,1024) = xs16 @ w_in16^T ; split xm / silu(z) ----
    wmma_gemm_kernel<0, 256, 4><<<(MROWS / 32) * (1024 / 64) / 8, 256, 0, stream>>>(
        xs16, w_in16, 1024, xm_pre, z16, nullptr, nullptr, nullptr);

    // ---- 4. depthwise conv + SiLU ----
    conv_silu_kernel<<<(MROWS * DINNER) / 256, 256, 0, stream>>>(
        xm_pre, conv_w, conv_b, xm_f32, xm16);

    // ---- 5. x-proj GEMM: dbl (M,48) = xm16 @ wx16^T  (NBLK=3 covers N=48) ----
    wmma_gemm_kernel<1, 512, 3><<<(MROWS / 32) * (48 / 48) / 8, 256, 0, stream>>>(
        xm16, wx16, 48, dbl, nullptr, nullptr, nullptr, nullptr);

    // ---- 6. delta = softplus(dt @ w_dt^T + b_dt) ----
    delta_kernel<<<(MROWS * DINNER) / 256, 256, 0, stream>>>(dbl, w_dt, b_dt, deltaB);

    // ---- 7. selective scan + gate ----
    scan_kernel<<<NB * 4, 128, 0, stream>>>(
        xm_f32, deltaB, dbl, a_log, d_skip, z16, ymul16);

    // ---- 8. out-proj GEMM + beta residual -> yhw (M,256) ----
    wmma_gemm_kernel<2, 512, 4><<<(MROWS / 32) * (256 / 64) / 8, 256, 0, stream>>>(
        ymul16, wout16, 256, yhw, nullptr, nullptr, beta, x);

    // ---- 9. FFN LN ----
    ln_ffn_kernel<<<MROWS / 8, 256, 0, stream>>>(yhw, ln_f_w, ln_f_b, ln16);

    // ---- 10. fc1 + leaky relu ----
    wmma_gemm_kernel<3, 256, 4><<<(MROWS / 32) * (512 / 64) / 8, 256, 0, stream>>>(
        ln16, fc116, 512, nullptr, t116, fc1_b, nullptr, nullptr);

    // ---- 11. fc2 + residual + transpose store to (B,256,H,W) ----
    wmma_gemm_kernel<4, 512, 4><<<(MROWS / 32) * (256 / 64) / 8, 256, 0, stream>>>(
        t116, fc216, 256, out, nullptr, fc2_b, nullptr, yhw);
}
